// ReachabilityGNN_43430709297337
// MI455X (gfx1250) — compile-verified, matrix operand order checked by reading, not run on hardware
//
#include <hip/hip_runtime.h>
#include <hip/hip_bf16.h>

#define NN 100000          // nodes
#define NE 1000000         // edges
#define NG 1024            // graphs
#define HID 64
#define HEADS 2
#define F2 128             // HEADS*HID
#define SLOPE 0.2f
#define TOT_E (NE + NN)    // edges + self loops

typedef __attribute__((ext_vector_type(16))) _Float16 v16h;
typedef __attribute__((ext_vector_type(8)))  _Float16 v8h;
typedef __attribute__((ext_vector_type(8)))  float    v8f;
typedef __attribute__((ext_vector_type(4)))  float    v4f;

static __device__ __forceinline__ float leaky(float v){ return v > 0.f ? v : SLOPE * v; }

// ---------------- utility kernels ----------------

__global__ void k_fill_f32(float* p, float v, int n){
  int i = blockIdx.x * blockDim.x + threadIdx.x;
  if (i < n) p[i] = v;
}

// transpose-convert f32 [rows,cols] -> f16 [cols,rows] (column of src contiguous in dst)
__global__ void k_transpose_f16(const float* __restrict__ src, _Float16* __restrict__ dst,
                                int rows, int cols){
  int i = blockIdx.x * blockDim.x + threadIdx.x;
  if (i >= rows * cols) return;
  int r = i / cols, c = i % cols;
  dst[c * rows + r] = (_Float16)src[r * cols + c];
}

// ---------------- GAT layer kernels ----------------

// conv1 projection: xp[n,128] = x[n,0:6] @ W1[6,128]
__global__ void k_lin1(const float* __restrict__ x, const float* __restrict__ W1,
                       float* __restrict__ xp){
  int i = blockIdx.x * blockDim.x + threadIdx.x;   // n*128 + j
  if (i >= NN * F2) return;
  int n = i >> 7, j = i & 127;
  float acc = 0.f;
  #pragma unroll
  for (int k = 0; k < 6; k++) acc += x[n * 6 + k] * W1[k * F2 + j];
  xp[i] = acc;
}

// per-(node,head) attention dots: a_s = <xp[n,h,:], att_src[h,:]>, a_d likewise
__global__ void k_att_dots(const float* __restrict__ xp, const float* __restrict__ as,
                           const float* __restrict__ ad, float* __restrict__ a_s,
                           float* __restrict__ a_d){
  int i = blockIdx.x * blockDim.x + threadIdx.x;   // n*2 + h
  if (i >= NN * HEADS) return;
  int n = i >> 1, h = i & 1;
  const float* row = xp + (size_t)n * F2 + h * HID;
  float s = 0.f, d = 0.f;
  #pragma unroll 8
  for (int c = 0; c < HID; c++){ s += row[c] * as[h * HID + c]; d += row[c] * ad[h * HID + c]; }
  a_s[i] = s; a_d[i] = d;
}

static __device__ __forceinline__ void edge_pair(const int* ei, int e, int& src, int& dst){
  if (e < NE){ src = ei[e]; dst = ei[NE + e]; }
  else       { src = dst = e - NE; }              // self loops appended
}

// segment-max of leakyrelu(a_s[src]+a_d[dst]) into emax[dst,h]  (emax pre-filled -inf)
__global__ void k_edge_max(const int* __restrict__ ei, const float* __restrict__ a_s,
                           const float* __restrict__ a_d, float* __restrict__ emax){
  int i = blockIdx.x * blockDim.x + threadIdx.x;   // e*2 + h
  if (i >= TOT_E * HEADS) return;
  int e = i >> 1, h = i & 1;
  int src, dst; edge_pair(ei, e, src, dst);
  float v = leaky(a_s[src * 2 + h] + a_d[dst * 2 + h]);
  int* addr = (int*)&emax[dst * 2 + h];
  if (v >= 0.f) atomicMax(addr, __float_as_int(v));
  else          atomicMin((unsigned int*)addr, (unsigned int)__float_as_int(v));
}

// ex = exp(e - emax[dst]);  store ex;  den[dst,h] += ex
__global__ void k_edge_exp(const int* __restrict__ ei, const float* __restrict__ a_s,
                           const float* __restrict__ a_d, const float* __restrict__ emax,
                           float* __restrict__ alpha, float* __restrict__ den){
  int i = blockIdx.x * blockDim.x + threadIdx.x;   // e*2 + h
  if (i >= TOT_E * HEADS) return;
  int e = i >> 1, h = i & 1;
  int src, dst; edge_pair(ei, e, src, dst);
  float v  = leaky(a_s[src * 2 + h] + a_d[dst * 2 + h]);
  float ex = __expf(v - emax[dst * 2 + h]);
  alpha[i] = ex;
  atomicAdd(&den[dst * 2 + h], ex);
}

// alpha = ex / (den[dst] + eps)
__global__ void k_alpha_norm(const int* __restrict__ ei, const float* __restrict__ den,
                             float* __restrict__ alpha){
  int i = blockIdx.x * blockDim.x + threadIdx.x;   // e*2 + h
  if (i >= TOT_E * HEADS) return;
  int e = i >> 1, h = i & 1;
  int src, dst; edge_pair(ei, e, src, dst);
  alpha[i] = alpha[i] / (den[dst * 2 + h] + 1e-16f);
}

// agg[dst,ch:ch+4] += xp[src,ch:ch+4] * alpha[e,h]
// One thread per (edge, 4-channel quad): one b128 load of xp + 4 f32 atomics.
// All traffic is L2-resident (xp/agg = 51 MB each << 192 MB L2).
__global__ void k_edge_msg4(const int* __restrict__ ei, const float* __restrict__ xp,
                            const float* __restrict__ alpha, float* __restrict__ agg){
  int idx = blockIdx.x * blockDim.x + threadIdx.x;  // e*32 + q  (q = quad index, 0..31)
  if (idx >= TOT_E * (F2 / 4)) return;
  int e = idx >> 5, q = idx & 31;
  int ch = q * 4, h = q >> 4;                       // h = (ch>>6)
  int src, dst; edge_pair(ei, e, src, dst);
  float a = alpha[e * 2 + h];
  v4f m = *(const v4f*)(xp + (size_t)src * F2 + ch);
  float* dp = agg + (size_t)dst * F2 + ch;
  atomicAdd(dp + 0, m.x * a);
  atomicAdd(dp + 1, m.y * a);
  atomicAdd(dp + 2, m.z * a);
  atomicAdd(dp + 3, m.w * a);
}

// head mean + bias + relu -> f16 row (dstStride lets us write into Z16's left half)
__global__ void k_head_mean_relu_f16(const float* __restrict__ agg, const float* __restrict__ bias,
                                     _Float16* __restrict__ dst, int dstStride){
  int i = blockIdx.x * blockDim.x + threadIdx.x;   // n*64 + c
  if (i >= NN * HID) return;
  int n = i >> 6, c = i & 63;
  float v = 0.5f * (agg[(size_t)n * F2 + c] + agg[(size_t)n * F2 + HID + c]) + bias[c];
  v = v > 0.f ? v : 0.f;
  dst[(size_t)n * dstStride + c] = (_Float16)v;
}

// ---------------- WMMA GEMM:  D[M,NC] = A[M,K] @ B[K,NC]  ----------------
// A row-major f16; B supplied as BT row-major [NC,K] f16 (i.e. columns contiguous).
// One wave per 16x16 D tile; K stepped by 32 via v_wmma_f32_16x16x32_f16.
template<int KTOT, bool EPI_BIAS_RELU>
__global__ void k_gemm_wmma(const _Float16* __restrict__ A, const _Float16* __restrict__ BT,
                            const float* __restrict__ bias, float* __restrict__ D,
                            int tilesN, int NC){
  int wave = (int)((blockIdx.x * blockDim.x + threadIdx.x) >> 5);
  int lane = threadIdx.x & 31;
  int tm = wave / tilesN, tn = wave % tilesN;
  int hi = lane >> 4;        // 0: K 0-7/16-23, 1: K 8-15/24-31 (ISA 16-bit A/B lane layout)
  int r  = lane & 15;        // row (A) / column (B) within tile
  const _Float16* arow = A  + (size_t)(tm * 16 + r) * KTOT;
  const _Float16* brow = BT + (size_t)(tn * 16 + r) * KTOT;
  v8f c = {};
  #pragma unroll
  for (int k0 = 0; k0 < KTOT; k0 += 32){
    union { v16h v; v8h h[2]; } a, b;
    a.h[0] = *(const v8h*)(arow + k0 +      hi * 8);
    a.h[1] = *(const v8h*)(arow + k0 + 16 + hi * 8);
    b.h[0] = *(const v8h*)(brow + k0 +      hi * 8);
    b.h[1] = *(const v8h*)(brow + k0 + 16 + hi * 8);
    c = __builtin_amdgcn_wmma_f32_16x16x32_f16(false, a.v, false, b.v, (short)0, c, false, false);
  }
  // D layout: lanes 0-15 -> rows 0-7 (vgpr i = row i), lanes 16-31 -> rows 8-15; col = lane&15
  int col   = tn * 16 + r;
  int rbase = tm * 16 + hi * 8;
  float bv = 0.f;
  if (EPI_BIAS_RELU) bv = bias[col];
  #pragma unroll
  for (int i = 0; i < 8; i++){
    float v = c[i];
    if (EPI_BIAS_RELU){ v += bv; v = v > 0.f ? v : 0.f; }
    D[(size_t)(rbase + i) * NC + col] = v;
  }
}

// ---------------- climber embed / concat / output head ----------------

__global__ void k_climber(const float* __restrict__ climber, const float* __restrict__ Wc,
                          const float* __restrict__ bc, _Float16* __restrict__ c16){
  int i = blockIdx.x * blockDim.x + threadIdx.x;   // g*64 + j
  if (i >= NG * HID) return;
  int g = i >> 6, j = i & 63;
  float acc = bc[j];
  #pragma unroll
  for (int k = 0; k < 4; k++) acc += climber[g * 4 + k] * Wc[k * HID + j];
  c16[i] = (_Float16)(acc > 0.f ? acc : 0.f);
}

__global__ void k_gather_climber(const int* __restrict__ batch, const _Float16* __restrict__ c16,
                                 _Float16* __restrict__ z16){
  int i = blockIdx.x * blockDim.x + threadIdx.x;   // n*64 + c
  if (i >= NN * HID) return;
  int n = i >> 6, c = i & 63;
  z16[(size_t)n * F2 + HID + c] = c16[batch[n] * HID + c];
}

__global__ void k_out(const float* __restrict__ zmid, const float* __restrict__ Wm2,
                      const float* __restrict__ bm2, float* __restrict__ out){
  int i = blockIdx.x * blockDim.x + threadIdx.x;   // n*4 + o
  if (i >= NN * 4) return;
  int n = i >> 2, o = i & 3;
  float acc = bm2[o];
  #pragma unroll 8
  for (int k = 0; k < HID; k++) acc += zmid[(size_t)n * HID + k] * Wm2[k * 4 + o];
  out[i] = acc;
}

// ---------------- host side ----------------

#define LB(n) (((n) + 255) / 256), 256, 0, stream

static void run_edge_phase(const int* ei, const float* xp, const float* a_s, const float* a_d,
                           float* emax, float* den, float* alpha, float* agg, hipStream_t stream){
  k_fill_f32<<<LB(NN * HEADS)>>>(emax, -__builtin_huge_valf(), NN * HEADS);
  k_fill_f32<<<LB(NN * HEADS)>>>(den, 0.f, NN * HEADS);
  k_fill_f32<<<LB(NN * F2)>>>(agg, 0.f, NN * F2);
  k_edge_max  <<<LB(TOT_E * HEADS)>>>(ei, a_s, a_d, emax);
  k_edge_exp  <<<LB(TOT_E * HEADS)>>>(ei, a_s, a_d, emax, alpha, den);
  k_alpha_norm<<<LB(TOT_E * HEADS)>>>(ei, den, alpha);
  k_edge_msg4 <<<LB(TOT_E * (F2 / 4))>>>(ei, xp, alpha, agg);
}

extern "C" void kernel_launch(void* const* d_in, const int* in_sizes, int n_in,
                              void* d_out, int out_size, void* d_ws, size_t ws_size,
                              hipStream_t stream){
  const float* x      = (const float*)d_in[0];
  const int*   ei     = (const int*)  d_in[1];
  const int*   batch  = (const int*)  d_in[2];
  const float* climber= (const float*)d_in[3];
  const float* W1     = (const float*)d_in[4];
  const float* as1    = (const float*)d_in[5];
  const float* ad1    = (const float*)d_in[6];
  const float* b1     = (const float*)d_in[7];
  const float* W2     = (const float*)d_in[8];
  const float* as2    = (const float*)d_in[9];
  const float* ad2    = (const float*)d_in[10];
  const float* b2     = (const float*)d_in[11];
  const float* Wc     = (const float*)d_in[12];
  const float* bc     = (const float*)d_in[13];
  const float* Wm1    = (const float*)d_in[14];
  const float* bm1    = (const float*)d_in[15];
  const float* Wm2    = (const float*)d_in[16];
  const float* bm2    = (const float*)d_in[17];
  float* out = (float*)d_out;

  // workspace carve-up (256B aligned)
  char* ws = (char*)d_ws;
  size_t off = 0;
  auto alloc = [&](size_t bytes){ size_t r = off; off += (bytes + 255) & ~(size_t)255; return r; };
  float*    XP    = (float*)   (ws + alloc((size_t)NN * F2 * 4));   // conv1 then conv2 features
  float*    AGG   = (float*)   (ws + alloc((size_t)NN * F2 * 4));   // aggregation; reused as ZMID
  float*    AS    = (float*)   (ws + alloc((size_t)NN * HEADS * 4));
  float*    AD    = (float*)   (ws + alloc((size_t)NN * HEADS * 4));
  float*    EMAX  = (float*)   (ws + alloc((size_t)NN * HEADS * 4));
  float*    DEN   = (float*)   (ws + alloc((size_t)NN * HEADS * 4));
  float*    ALPHA = (float*)   (ws + alloc((size_t)TOT_E * HEADS * 4));
  _Float16* H16   = (_Float16*)(ws + alloc((size_t)NN * HID * 2));
  _Float16* Z16   = (_Float16*)(ws + alloc((size_t)NN * F2 * 2));
  _Float16* C16   = (_Float16*)(ws + alloc((size_t)NG * HID * 2));
  _Float16* W2T   = (_Float16*)(ws + alloc((size_t)F2 * HID * 2));  // [128 cols][64 K]
  _Float16* WM1T  = (_Float16*)(ws + alloc((size_t)HID * F2 * 2));  // [64 cols][128 K]
  float*    ZMID  = AGG;                                            // [NN,64] after conv2 done

  // weight prep (tiny, every call)
  k_transpose_f16<<<LB(HID * F2)>>>(W2,  W2T,  HID, F2);   // W2 [64,128] -> [128,64]
  k_transpose_f16<<<LB(F2 * HID)>>>(Wm1, WM1T, F2, HID);   // Wm1 [128,64] -> [64,128]
  k_climber<<<LB(NG * HID)>>>(climber, Wc, bc, C16);

  // ---- conv1 ----
  k_lin1<<<LB(NN * F2)>>>(x, W1, XP);
  k_att_dots<<<LB(NN * HEADS)>>>(XP, as1, ad1, AS, AD);
  run_edge_phase(ei, XP, AS, AD, EMAX, DEN, ALPHA, AGG, stream);
  k_head_mean_relu_f16<<<LB(NN * HID)>>>(AGG, b1, H16, HID);

  // ---- conv2 ----  xp2 = h1 @ W2 via WMMA  (6250*8 waves, 4 waves/block, exact fill)
  k_gemm_wmma<HID, false><<<(NN / 16) * (F2 / 16) / 4, 128, 0, stream>>>(H16, W2T, nullptr, XP, F2 / 16, F2);
  k_att_dots<<<LB(NN * HEADS)>>>(XP, as2, ad2, AS, AD);
  run_edge_phase(ei, XP, AS, AD, EMAX, DEN, ALPHA, AGG, stream);
  k_head_mean_relu_f16<<<LB(NN * HID)>>>(AGG, b2, Z16, F2);         // left half of z
  k_gather_climber<<<LB(NN * HID)>>>(batch, C16, Z16);              // right half of z

  // ---- MLP ----  zmid = relu(z @ Wm1 + bm1) via WMMA (bias+relu epilogue), then out head
  k_gemm_wmma<F2, true><<<(NN / 16) * (HID / 16) / 4, 128, 0, stream>>>(Z16, WM1T, bm1, ZMID, HID / 16, HID);
  k_out<<<LB(NN * 4)>>>(ZMID, Wm2, bm2, out);

  (void)in_sizes; (void)n_in; (void)out_size; (void)ws_size;
}
